// SRG_no_GRU_37838661877776
// MI455X (gfx1250) — matrix operand order repaired and checked by reading, not compiled, not run on hardware
//
#include <hip/hip_runtime.h>
#include <hip/hip_bf16.h>

typedef __attribute__((ext_vector_type(16))) _Float16 v16h;
typedef __attribute__((ext_vector_type(2)))  _Float16 h2v;
typedef __attribute__((ext_vector_type(8)))  float    v8f;

namespace {

// ---- monotonic float <-> uint key for atomicMax (segment_max) ----
__device__ inline unsigned fkey(float f) {
  unsigned u = __float_as_uint(f);
  return (u & 0x80000000u) ? ~u : (u | 0x80000000u);
}
__device__ inline float funkey(unsigned k) {
  unsigned u = (k & 0x80000000u) ? (k & 0x7fffffffu) : ~k;
  return __uint_as_float(u);
}

__global__ void k_zero(float* __restrict__ p, int n) {
  int i = blockIdx.x * blockDim.x + threadIdx.x;
  if (i < n) p[i] = 0.f;
}

__global__ void k_degree(const int* __restrict__ s, const int* __restrict__ d,
                         float* __restrict__ degs, float* __restrict__ degd, int E) {
  int i = blockIdx.x * blockDim.x + threadIdx.x;
  if (i >= E) return;
  atomicAdd(&degs[s[i]], 1.f);
  atomicAdd(&degd[d[i]], 1.f);
}

__global__ void k_rsqrt_clip(float* __restrict__ p, int n) {
  int i = blockIdx.x * blockDim.x + threadIdx.x;
  if (i < n) p[i] = rsqrtf(fmaxf(p[i], 1.f));
}

// out[row][:] = x[row][:] * r[row]   (128 cols, float4 per thread)
__global__ void k_scale_rows(const float* __restrict__ x, const float* __restrict__ r,
                             float* __restrict__ o, int n) {
  int t = blockIdx.x * blockDim.x + threadIdx.x;
  if (t >= n * 32) return;
  int row = t >> 5;
  float4 v = ((const float4*)x)[t];
  float s = r[row];
  v.x *= s; v.y *= s; v.z *= s; v.w *= s;
  ((float4*)o)[t] = v;
}

// agg[dst[e]][:] += h[src[e]][:]   (128 cols, float4 chunk per thread)
__global__ void k_scatter_add(const float* __restrict__ h, const int* __restrict__ src,
                              const int* __restrict__ dst, float* __restrict__ agg, int E) {
  int t = blockIdx.x * blockDim.x + threadIdx.x;
  if (t >= E * 32) return;
  int e = t >> 5, c4 = t & 31;
  float4 v = *(const float4*)(h + (size_t)src[e] * 128 + c4 * 4);
  float* o = agg + (size_t)dst[e] * 128 + c4 * 4;
  atomicAdd(o + 0, v.x); atomicAdd(o + 1, v.y);
  atomicAdd(o + 2, v.z); atomicAdd(o + 3, v.w);
}

// h1 = relu(agg * r_in[row] + b[col])
__global__ void k_gcn_fin(const float* __restrict__ agg, const float* __restrict__ rin,
                          const float* __restrict__ b, float* __restrict__ out, int n) {
  int t = blockIdx.x * blockDim.x + threadIdx.x;
  if (t >= n * 128) return;
  int row = t >> 7, c = t & 127;
  out[t] = fmaxf(agg[t] * rin[row] + b[c], 0.f);
}

// el[n,h] = h[n,h*32:]·al[h,:], er likewise
__global__ void k_attn_coef(const float* __restrict__ h, const float* __restrict__ al,
                            const float* __restrict__ ar, float* __restrict__ el,
                            float* __restrict__ er, int n) {
  int t = blockIdx.x * blockDim.x + threadIdx.x;
  if (t >= n * 4) return;
  int node = t >> 2, hd = t & 3;
  const float* hp = h + (size_t)node * 128 + hd * 32;
  const float* alp = al + hd * 32;
  const float* arp = ar + hd * 32;
  float sl = 0.f, sr = 0.f;
  #pragma unroll 8
  for (int d = 0; d < 32; ++d) { float hv = hp[d]; sl += hv * alp[d]; sr += hv * arp[d]; }
  el[t] = sl; er[t] = sr;
}

// e = leaky_relu(el[src]+er[dst]); store; segment_max into mmax
__global__ void k_edge_att(const float* __restrict__ el, const float* __restrict__ er,
                           const int* __restrict__ src, const int* __restrict__ dst,
                           float* __restrict__ eatt, unsigned* __restrict__ mmax, int E) {
  int t = blockIdx.x * blockDim.x + threadIdx.x;
  if (t >= E * 4) return;
  int e = t >> 2, hd = t & 3;
  float x = el[src[e] * 4 + hd] + er[dst[e] * 4 + hd];
  x = x > 0.f ? x : 0.2f * x;
  eatt[t] = x;
  atomicMax(&mmax[dst[e] * 4 + hd], fkey(x));
}

__global__ void k_edge_denom(const float* __restrict__ eatt, const unsigned* __restrict__ mmax,
                             const int* __restrict__ dst, float* __restrict__ denom, int E) {
  int t = blockIdx.x * blockDim.x + threadIdx.x;
  if (t >= E * 4) return;
  int e = t >> 2, hd = t & 3;
  int di = dst[e] * 4 + hd;
  atomicAdd(&denom[di], expf(eatt[t] - funkey(mmax[di])));
}

// alpha (in place over eatt)
__global__ void k_edge_alpha(float* __restrict__ eatt, const unsigned* __restrict__ mmax,
                             const float* __restrict__ denom, const int* __restrict__ dst, int E) {
  int t = blockIdx.x * blockDim.x + threadIdx.x;
  if (t >= E * 4) return;
  int e = t >> 2, hd = t & 3;
  int di = dst[e] * 4 + hd;
  eatt[t] = expf(eatt[t] - funkey(mmax[di])) / denom[di];
}

// agg[dst] += alpha[e,head] * h[src]   (float4 chunk; head = c4/8)
__global__ void k_gat_agg(const float* __restrict__ h, const float* __restrict__ alpha,
                          const int* __restrict__ src, const int* __restrict__ dst,
                          float* __restrict__ agg, int E) {
  int t = blockIdx.x * blockDim.x + threadIdx.x;
  if (t >= E * 32) return;
  int e = t >> 5, c4 = t & 31, hd = c4 >> 3;
  float a = alpha[e * 4 + hd];
  float4 v = *(const float4*)(h + (size_t)src[e] * 128 + c4 * 4);
  float* o = agg + (size_t)dst[e] * 128 + c4 * 4;
  atomicAdd(o + 0, v.x * a); atomicAdd(o + 1, v.y * a);
  atomicAdd(o + 2, v.z * a); atomicAdd(o + 3, v.w * a);
}

// z[(n*3+mp)*128+c] = elu(agg[n*128+c] + b[c])
__global__ void k_gat_fin(const float* __restrict__ agg, const float* __restrict__ b,
                          float* __restrict__ z, int mp, int n) {
  int t = blockIdx.x * blockDim.x + threadIdx.x;
  if (t >= n * 128) return;
  int node = t >> 7, c = t & 127;
  float x = agg[t] + b[c];
  x = x > 0.f ? x : (expf(x) - 1.f);
  z[((size_t)node * 3 + mp) * 128 + c] = x;
}

// beta = softmax over 3 metapaths; h2 = sum beta*z
__global__ void k_sem_combine(const float* __restrict__ z, const float* __restrict__ w,
                              float* __restrict__ h2, int n) {
  int t = blockIdx.x * blockDim.x + threadIdx.x;
  if (t >= n * 32) return;
  int node = t >> 5, c4 = t & 31;
  float w0 = w[node * 3 + 0], w1 = w[node * 3 + 1], w2 = w[node * 3 + 2];
  float mx = fmaxf(w0, fmaxf(w1, w2));
  float e0 = expf(w0 - mx), e1 = expf(w1 - mx), e2 = expf(w2 - mx);
  float inv = 1.f / (e0 + e1 + e2);
  float b0 = e0 * inv, b1 = e1 * inv, b2 = e2 * inv;
  float4 z0 = *(const float4*)(z + ((size_t)node * 3 + 0) * 128 + c4 * 4);
  float4 z1 = *(const float4*)(z + ((size_t)node * 3 + 1) * 128 + c4 * 4);
  float4 z2 = *(const float4*)(z + ((size_t)node * 3 + 2) * 128 + c4 * 4);
  float4 o;
  o.x = b0 * z0.x + b1 * z1.x + b2 * z2.x;
  o.y = b0 * z0.y + b1 * z1.y + b2 * z2.y;
  o.z = b0 * z0.z + b1 * z1.z + b2 * z2.z;
  o.w = b0 * z0.w + b1 * z1.w + b2 * z2.w;
  *(float4*)(h2 + (size_t)node * 128 + c4 * 4) = o;
}

__global__ void k_add_scalar(float* __restrict__ o, const float* __restrict__ b, int n) {
  int i = blockIdx.x * blockDim.x + threadIdx.x;
  if (i < n) o[i] += b[0];
}

// ---------------- LDS-staged WMMA GEMM ----------------
// Block = TN waves (32*TN threads); covers 16 rows x 16*TN cols of out.
// grid.x = rows/16, grid.y = ncols/(16*TN). All tile dims divide exactly.
// Per 32-wide K-step: stage A strip (16x32) and B slab (32 x 16*TN) into LDS
// as f16 (rows padded to 34 halves = 17 dwords -> conflict-free), then each
// wave assembles fragments with 32-bit LDS reads and issues
// v_wmma_f32_16x16x32_f16.
// AMODE 0: dense A (lda=K). AMODE 1: A row r = concat[h1[u_r], h2[u_r], h2[v_r]] (128 each).
// EPI 0: plain store. EPI 1: atomicAdd(out[row], tanh(c+bias[col])*w2[col]).
//        EPI 2: atomicAdd(out[row], elu(c+bias[col])*w2[col]).
#define LPAD 34
template <int TN, int AMODE, int EPI>
__global__ __launch_bounds__(32 * TN) void k_gemm(
    const float* __restrict__ A, const float* __restrict__ B, float* __restrict__ out,
    int rows, int K, int ncols,
    const float* __restrict__ bias, const float* __restrict__ w2,
    const float* __restrict__ h1, const float* __restrict__ h2,
    const int* __restrict__ pairs) {
  constexpr int NT = 32 * TN;
  __shared__ _Float16 lA[16 * LPAD];
  __shared__ _Float16 lB[16 * TN * LPAD];
  __shared__ int lu[16], lv[16];

  const int tid = threadIdx.x;
  const int row0 = blockIdx.x * 16;
  const int col0 = blockIdx.y * 16 * TN;
  const int wid  = tid >> 5;                 // wave id = column tile within block
  const int lane = tid & 31;
  const int hw = lane >> 4;                  // half-wave
  const int ln = lane & 15;

  if (AMODE == 1) {
    if (tid < 16) {
      lu[tid] = pairs[2 * (row0 + tid) + 0];
      lv[tid] = pairs[2 * (row0 + tid) + 1];
    }
  }
  __syncthreads();

  v8f c = {};
  for (int kk = 0; kk < K; kk += 32) {
    // ---- stage A: 16 rows x 32 k  (256 float2, coalesced) ----
    #pragma unroll
    for (int f = tid; f < 256; f += NT) {
      int row = f >> 4, pr = f & 15;         // k = 2*pr
      const float* ap;
      if (AMODE == 0) {
        ap = A + (size_t)(row0 + row) * K + kk;
      } else {
        int seg = kk >> 7;                   // 0: h1[u], 1: h2[u], 2: h2[v]
        int idx = (seg == 2) ? lv[row] : lu[row];
        ap = ((seg == 0) ? h1 : h2) + (size_t)idx * 128 + (kk & 127);
      }
      float2 g = *(const float2*)(ap + 2 * pr);
      h2v hv; hv.x = (_Float16)g.x; hv.y = (_Float16)g.y;
      *(h2v*)(lA + row * LPAD + 2 * pr) = hv;
    }
    // ---- stage B transposed: lB[col][k], 32 k-rows x 16*TN cols (float4, coalesced) ----
    #pragma unroll
    for (int f = tid; f < 128 * TN; f += NT) {
      int k = f / (4 * TN), c4 = f % (4 * TN);
      float4 g = *(const float4*)(B + (size_t)(kk + k) * ncols + col0 + 4 * c4);
      lB[(4 * c4 + 0) * LPAD + k] = (_Float16)g.x;
      lB[(4 * c4 + 1) * LPAD + k] = (_Float16)g.y;
      lB[(4 * c4 + 2) * LPAD + k] = (_Float16)g.z;
      lB[(4 * c4 + 3) * LPAD + k] = (_Float16)g.w;
    }
    __syncthreads();

    // ---- fragments from LDS ----
    v16h a, b;
    const _Float16* pa = lA + ln * LPAD;            // A row = ln
    const _Float16* pb = lB + (wid * 16 + ln) * LPAD; // B col = wid*16+ln
    #pragma unroll
    for (int v = 0; v < 8; ++v) {
      // A: k = 16*(v/4) + 8*half + 2*(v%4) + {0,1}   (ISA 16-bit A 16x32 layout)
      int ka = 16 * (v >> 2) + 8 * hw + 2 * (v & 3);
      h2v xa = *(const h2v*)(pa + ka);
      a[2 * v + 0] = xa.x; a[2 * v + 1] = xa.y;
      // B: k = 16*(v/4) + 8*half + 2*(v%4) + {0,1}   (same k order as A)
      int kb = 16 * (v >> 2) + 8 * hw + 2 * (v & 3);
      h2v xb = *(const h2v*)(pb + kb);
      b[2 * v + 0] = xb.x; b[2 * v + 1] = xb.y;
    }
    c = __builtin_amdgcn_wmma_f32_16x16x32_f16(false, a, false, b, (short)0, c, false, false);
    __syncthreads();
  }

  int col = col0 + wid * 16 + ln;
  #pragma unroll
  for (int r = 0; r < 8; ++r) {
    int row = row0 + r + 8 * hw;             // C/D layout: VGPR r -> M=r (+8 upper half)
    float val = c[r];
    if (EPI == 0) {
      out[(size_t)row * ncols + col] = val;
    } else if (EPI == 1) {
      atomicAdd(&out[row], tanhf(val + bias[col]) * w2[col]);
    } else {
      float hh = val + bias[col];
      hh = hh > 0.f ? hh : (expf(hh) - 1.f);
      atomicAdd(&out[row], hh * w2[col]);
    }
  }
}

} // namespace

extern "C" void kernel_launch(void* const* d_in, const int* in_sizes, int n_in,
                              void* d_out, int out_size, void* d_ws, size_t ws_size,
                              hipStream_t stream) {
  const float* feat1  = (const float*)d_in[0];
  const float* feat2  = (const float*)d_in[1];
  const int*   src1   = (const int*)d_in[2];
  const int*   dst1   = (const int*)d_in[3];
  const int*   src_mp = (const int*)d_in[4];
  const int*   dst_mp = (const int*)d_in[5];
  const int*   pairs  = (const int*)d_in[6];
  const float* gcn_W0 = (const float*)d_in[7];
  const float* gcn_b0 = (const float*)d_in[8];
  const float* gcn_W1 = (const float*)d_in[9];
  const float* gcn_b1 = (const float*)d_in[10];
  const float* gat_W  = (const float*)d_in[11];
  const float* gat_al = (const float*)d_in[12];
  const float* gat_ar = (const float*)d_in[13];
  const float* gat_b  = (const float*)d_in[14];
  const float* sem_W1 = (const float*)d_in[15];
  const float* sem_b1 = (const float*)d_in[16];
  const float* sem_W2 = (const float*)d_in[17];
  const float* fc_W1  = (const float*)d_in[18];
  const float* fc_b1  = (const float*)d_in[19];
  const float* fc_W2  = (const float*)d_in[20];
  const float* fc_b2  = (const float*)d_in[21];
  float* outp = (float*)d_out;

  const int N = in_sizes[0] / 128;
  const int E = in_sizes[2];
  const int P = in_sizes[6] / 2;
  (void)n_in; (void)out_size; (void)ws_size;

  float* ws = (float*)d_ws;
  size_t off = 0;
  auto take = [&](size_t n) { float* p = ws + off; off += n; return p; };
  float* r_out = take(N);
  float* r_in  = take(N);
  float* xs    = take((size_t)N * 128);
  float* hbuf  = take((size_t)N * 128);
  float* agg   = take((size_t)N * 128);
  float* h1    = take((size_t)N * 128);
  float* zbuf  = take((size_t)N * 3 * 128);
  float* h2    = take((size_t)N * 128);
  float* el    = take((size_t)N * 4);
  float* er    = take((size_t)N * 4);
  float* eatt  = take((size_t)E * 4);
  float* mmaxf = take((size_t)N * 4);
  float* denom = take((size_t)N * 4);
  float* wsem  = take((size_t)N * 3);

  dim3 B256(256);
  auto nb = [](long n) { return dim3((unsigned)((n + 255) / 256)); };
  const float* nf = nullptr;
  const int* ni = nullptr;

  // ---------- GCN degrees ----------
  hipLaunchKernelGGL(k_zero, nb(N), B256, 0, stream, r_out, N);
  hipLaunchKernelGGL(k_zero, nb(N), B256, 0, stream, r_in, N);
  hipLaunchKernelGGL(k_degree, nb(E), B256, 0, stream, src1, dst1, r_out, r_in, E);
  hipLaunchKernelGGL(k_rsqrt_clip, nb(N), B256, 0, stream, r_out, N);
  hipLaunchKernelGGL(k_rsqrt_clip, nb(N), B256, 0, stream, r_in, N);

  // ---------- GCN layer 0 ----------
  hipLaunchKernelGGL(k_scale_rows, nb((long)N * 32), B256, 0, stream, feat1, r_out, xs, N);
  hipLaunchKernelGGL((k_gemm<8, 0, 0>), dim3(N / 16, 1), dim3(256), 0, stream,
                     xs, gcn_W0, hbuf, N, 128, 128, nf, nf, nf, nf, ni);
  hipLaunchKernelGGL(k_zero, nb((long)N * 128), B256, 0, stream, agg, N * 128);
  hipLaunchKernelGGL(k_scatter_add, nb((long)E * 32), B256, 0, stream, hbuf, src1, dst1, agg, E);
  hipLaunchKernelGGL(k_gcn_fin, nb((long)N * 128), B256, 0, stream, agg, r_in, gcn_b0, h1, N);

  // ---------- GCN layer 1 ----------
  hipLaunchKernelGGL(k_scale_rows, nb((long)N * 32), B256, 0, stream, h1, r_out, xs, N);
  hipLaunchKernelGGL((k_gemm<8, 0, 0>), dim3(N / 16, 1), dim3(256), 0, stream,
                     xs, gcn_W1, hbuf, N, 128, 128, nf, nf, nf, nf, ni);
  hipLaunchKernelGGL(k_zero, nb((long)N * 128), B256, 0, stream, agg, N * 128);
  hipLaunchKernelGGL(k_scatter_add, nb((long)E * 32), B256, 0, stream, hbuf, src1, dst1, agg, E);
  hipLaunchKernelGGL(k_gcn_fin, nb((long)N * 128), B256, 0, stream, agg, r_in, gcn_b1, h1, N);

  // ---------- GAT per metapath ----------
  for (int i = 0; i < 3; ++i) {
    const int* sp = src_mp + (size_t)i * E;
    const int* dp = dst_mp + (size_t)i * E;
    hipLaunchKernelGGL((k_gemm<8, 0, 0>), dim3(N / 16, 1), dim3(256), 0, stream,
                       feat2, gat_W + (size_t)i * 128 * 128, hbuf, N, 128, 128,
                       nf, nf, nf, nf, ni);
    hipLaunchKernelGGL(k_attn_coef, nb((long)N * 4), B256, 0, stream,
                       hbuf, gat_al + (size_t)i * 4 * 32, gat_ar + (size_t)i * 4 * 32, el, er, N);
    hipLaunchKernelGGL(k_zero, nb((long)N * 4), B256, 0, stream, mmaxf, N * 4); // key 0 == -inf
    hipLaunchKernelGGL(k_edge_att, nb((long)E * 4), B256, 0, stream,
                       el, er, sp, dp, eatt, (unsigned*)mmaxf, E);
    hipLaunchKernelGGL(k_zero, nb((long)N * 4), B256, 0, stream, denom, N * 4);
    hipLaunchKernelGGL(k_edge_denom, nb((long)E * 4), B256, 0, stream,
                       eatt, (const unsigned*)mmaxf, dp, denom, E);
    hipLaunchKernelGGL(k_edge_alpha, nb((long)E * 4), B256, 0, stream,
                       eatt, (const unsigned*)mmaxf, denom, dp, E);
    hipLaunchKernelGGL(k_zero, nb((long)N * 128), B256, 0, stream, agg, N * 128);
    hipLaunchKernelGGL(k_gat_agg, nb((long)E * 32), B256, 0, stream, hbuf, eatt, sp, dp, agg, E);
    hipLaunchKernelGGL(k_gat_fin, nb((long)N * 128), B256, 0, stream,
                       agg, gat_b + (size_t)i * 128, zbuf, i, N);
  }

  // ---------- Semantic attention: w = tanh(z@W1+b1)@W2 fused in GEMM epilogue ----------
  hipLaunchKernelGGL(k_zero, nb((long)N * 3), B256, 0, stream, wsem, N * 3);
  hipLaunchKernelGGL((k_gemm<4, 0, 1>), dim3((N * 3) / 16, 1), dim3(128), 0, stream,
                     zbuf, sem_W1, wsem, N * 3, 128, 64, sem_b1, sem_W2, nf, nf, ni);
  hipLaunchKernelGGL(k_sem_combine, nb((long)N * 32), B256, 0, stream, zbuf, wsem, h2, N);

  // ---------- Pair MLP: gathered-concat A, FC1(elu) + FC2 dot fused ----------
  hipLaunchKernelGGL(k_zero, nb(P), B256, 0, stream, outp, P);
  hipLaunchKernelGGL((k_gemm<8, 1, 2>), dim3(P / 16, 2), dim3(256), 0, stream,
                     nf, fc_W1, outp, P, 384, 256, fc_b1, fc_W2, h1, h2, pairs);
  hipLaunchKernelGGL(k_add_scalar, nb(P), B256, 0, stream, outp, fc_b2, P);
}
